// ItemConv_51041391345937
// MI455X (gfx1250) — matrix operand order repaired and checked by reading, not compiled, not run on hardware
//
#include <hip/hip_runtime.h>
#include <hip/hip_bf16.h>

typedef __attribute__((ext_vector_type(16))) _Float16 v16h;
typedef __attribute__((ext_vector_type(4)))  _Float16 v4h;
typedef __attribute__((ext_vector_type(8)))  float    v8f;
typedef __attribute__((ext_vector_type(4)))  float    v4f;

#define EMB 128

// ---------------------------------------------------------------------------
// GEMM: T = H @ W^T   (H: [n,128] f32, W: [128,128] f32, T: [n,128] f32)
// Wave32 WMMA f32_16x16x32_f16. One wave computes a 16-row M tile across all
// 128 output columns. W is converted to f16 and staged in LDS (32 KB).
// ---------------------------------------------------------------------------
__launch_bounds__(256)
__global__ void itemconv_gemm_wT(const float* __restrict__ H,
                                 const float* __restrict__ W,
                                 float* __restrict__ T, int nrows) {
    __shared__ _Float16 Wlds[EMB * EMB];   // row n, col k : W[n][k] (32 KB)

    // cooperative f32 -> f16 conversion of the weight matrix into LDS
    for (int i = threadIdx.x; i < EMB * EMB / 4; i += 256) {
        const v4f w = ((const v4f*)W)[i];
        v4h hv;
        hv[0] = (_Float16)w[0]; hv[1] = (_Float16)w[1];
        hv[2] = (_Float16)w[2]; hv[3] = (_Float16)w[3];
        *(v4h*)(&Wlds[i * 4]) = hv;        // ds_store_b64
    }
    __syncthreads();

    const int wave = threadIdx.x >> 5;          // 0..7
    const int lane = threadIdx.x & 31;
    const int tile = blockIdx.x * 8 + wave;     // 16-row tile index
    if (tile * 16 >= nrows) return;             // nrows % 16 == 0 here

    const int hi = lane >> 4;                   // lane group 0 / 1
    const int m  = tile * 16 + (lane & 15);
    const float* hrow = H + (size_t)m * EMB;

    // A fragments for all 4 K-blocks (this code's contiguous k order, matching B):
    //  lanes 0-15 : K run [kb*32+0..15]
    //  lanes16-31 : K run [kb*32+16..31]
    v16h a[4];
#pragma unroll
    for (int kb = 0; kb < 4; ++kb) {
        const int k0 = kb * 32 + hi * 16;
        const v4f f0 = *(const v4f*)(hrow + k0);
        const v4f f1 = *(const v4f*)(hrow + k0 + 4);
        const v4f f2 = *(const v4f*)(hrow + k0 + 8);
        const v4f f3 = *(const v4f*)(hrow + k0 + 12);
#pragma unroll
        for (int j = 0; j < 4; ++j) {
            a[kb][j]      = (_Float16)f0[j];
            a[kb][4 + j]  = (_Float16)f1[j];
            a[kb][8 + j]  = (_Float16)f2[j];
            a[kb][12 + j] = (_Float16)f3[j];
        }
    }

    // Loop over the 8 output-column tiles; chain 4 WMMAs over K per tile.
#pragma unroll
    for (int nt = 0; nt < 8; ++nt) {
        v8f acc = {};
        const int nrow = nt * 16 + (lane & 15);   // output column = W row
#pragma unroll
        for (int kb = 0; kb < 4; ++kb) {
            // B, same contiguous k order as A: lane group selects K half, 16 contiguous
            // halves per lane -> two ds_load_b128
            const int kk = kb * 32 + hi * 16;
            const v16h b = *(const v16h*)(&Wlds[nrow * EMB + kk]);
            acc = __builtin_amdgcn_wmma_f32_16x16x32_f16(
                false, a[kb], false, b, (short)0, acc, false, false);
        }
        // D layout: VGPR v -> row tile*16 + 8*hi + v, col nt*16 + (lane&15)
        float* out = T + (size_t)(tile * 16 + 8 * hi) * EMB + nt * 16 + (lane & 15);
#pragma unroll
        for (int v = 0; v < 8; ++v) out[(size_t)v * EMB] = acc[v];
    }
}

// ---------------------------------------------------------------------------
// SpMM scatter: O[rows[e]] += vals[e] * T[cols[e]]   (one wave per edge)
// 512 B contiguous gather per edge; 4 native fp32 global atomic adds per lane.
// ---------------------------------------------------------------------------
__launch_bounds__(256)
__global__ void itemconv_spmm(const float* __restrict__ T,
                              const float* __restrict__ vals,
                              const int* __restrict__ rows,
                              const int* __restrict__ cols,
                              float* __restrict__ O, int nedges) {
    const int gid = blockIdx.x * blockDim.x + threadIdx.x;
    const int e = gid >> 5;
    if (e >= nedges) return;
    const int lane = threadIdx.x & 31;

    const float v = vals[e];
    const int   c = cols[e];
    const int   r = rows[e];

    const v4f src = *(const v4f*)(T + (size_t)c * EMB + lane * 4);
    float* dst = O + (size_t)r * EMB + lane * 4;

    const float x0 = v * src[0], x1 = v * src[1], x2 = v * src[2], x3 = v * src[3];
    asm volatile("global_atomic_add_f32 %0, %1, off" :: "v"(dst + 0), "v"(x0) : "memory");
    asm volatile("global_atomic_add_f32 %0, %1, off" :: "v"(dst + 1), "v"(x1) : "memory");
    asm volatile("global_atomic_add_f32 %0, %1, off" :: "v"(dst + 2), "v"(x2) : "memory");
    asm volatile("global_atomic_add_f32 %0, %1, off" :: "v"(dst + 3), "v"(x3) : "memory");
}

// ---------------------------------------------------------------------------
// Row L2-norm + accumulate: acc[row] += h[row] / max(||h[row]||, 1e-12)
// One wave per row; wave32 shuffle reduction.
// ---------------------------------------------------------------------------
__launch_bounds__(256)
__global__ void itemconv_norm_acc(const float* __restrict__ Hn,
                                  float* __restrict__ acc, int nrows) {
    const int wave = threadIdx.x >> 5;
    const int lane = threadIdx.x & 31;
    const int row = blockIdx.x * 8 + wave;
    if (row >= nrows) return;

    const v4f hv = *(const v4f*)(Hn + (size_t)row * EMB + lane * 4);
    float s = hv[0] * hv[0] + hv[1] * hv[1] + hv[2] * hv[2] + hv[3] * hv[3];
#pragma unroll
    for (int off = 16; off >= 1; off >>= 1) s += __shfl_xor(s, off, 32);

    const float inv = 1.0f / fmaxf(sqrtf(s), 1e-12f);
    v4f* ap = (v4f*)(acc + (size_t)row * EMB + lane * 4);
    v4f av = *ap;
    av[0] += hv[0] * inv; av[1] += hv[1] * inv;
    av[2] += hv[2] * inv; av[3] += hv[3] * inv;
    *ap = av;
}

// ---------------------------------------------------------------------------
// Utility kernels
// ---------------------------------------------------------------------------
__global__ void itemconv_init(const float* __restrict__ E,
                              float* __restrict__ h,
                              float* __restrict__ acc, int n4) {
    const int i = blockIdx.x * blockDim.x + threadIdx.x;
    if (i < n4) {
        const v4f v = ((const v4f*)E)[i];
        ((v4f*)h)[i] = v;
        ((v4f*)acc)[i] = v;
    }
}

__global__ void itemconv_zero(float* __restrict__ p, int n4) {
    const int i = blockIdx.x * blockDim.x + threadIdx.x;
    if (i < n4) { v4f z = {}; ((v4f*)p)[i] = z; }
}

__global__ void itemconv_scale(float* __restrict__ p, float s, int n4) {
    const int i = blockIdx.x * blockDim.x + threadIdx.x;
    if (i < n4) {
        v4f v = ((v4f*)p)[i];
        v[0] *= s; v[1] *= s; v[2] *= s; v[3] *= s;
        ((v4f*)p)[i] = v;
    }
}

// ---------------------------------------------------------------------------
// Launch: inputs = embedding [N*128], adj_vals [E], weights [3*128*128],
//         rows [E], cols [E].  Output = acc/(L+1), [N*128] f32.
// Workspace: h_cur (N*128 f32) + t (N*128 f32) = 102.4 MB.
// ---------------------------------------------------------------------------
extern "C" void kernel_launch(void* const* d_in, const int* in_sizes, int n_in,
                              void* d_out, int out_size, void* d_ws, size_t ws_size,
                              hipStream_t stream) {
    const float* emb   = (const float*)d_in[0];
    const float* vals  = (const float*)d_in[1];
    const float* wts   = (const float*)d_in[2];
    const int*   rows  = (const int*)d_in[3];
    const int*   cols  = (const int*)d_in[4];
    float* acc = (float*)d_out;

    const int N = in_sizes[0] / EMB;      // 100000
    const int E = in_sizes[1];            // 1600000
    const int L = in_sizes[2] / (EMB * EMB);  // 3

    float* h_cur = (float*)d_ws;
    float* t     = h_cur + (size_t)N * EMB;

    const int n4 = N * EMB / 4;
    const int thr = 256;

    itemconv_init<<<(n4 + thr - 1) / thr, thr, 0, stream>>>(emb, h_cur, acc, n4);

    const int ntiles = (N + 15) / 16;
    const int gemm_blocks = (ntiles + 7) / 8;
    const int spmm_blocks = (E * 32 + thr - 1) / thr;
    const int row_blocks  = (N + 7) / 8;

    for (int i = 0; i < L; ++i) {
        const float* W = wts + (size_t)i * EMB * EMB;
        itemconv_gemm_wT<<<gemm_blocks, thr, 0, stream>>>(h_cur, W, t, N);
        itemconv_zero<<<(n4 + thr - 1) / thr, thr, 0, stream>>>(h_cur, n4);
        itemconv_spmm<<<spmm_blocks, thr, 0, stream>>>(t, vals, rows, cols, h_cur, E);
        itemconv_norm_acc<<<row_blocks, thr, 0, stream>>>(h_cur, acc, N);
    }

    itemconv_scale<<<(n4 + thr - 1) / thr, thr, 0, stream>>>(acc, 1.0f / (L + 1), n4);
}
